// SelfAttention_26207890440322
// MI455X (gfx1250) — matrix operand order repaired and checked by reading, not compiled, not run on hardware
//
#include <hip/hip_runtime.h>
#include <hip/hip_bf16.h>
#include <math.h>

// ---------------------------------------------------------------------------
// Self-attention for MI455X (gfx1250, wave32, WMMA).
//   q = x@Wq^T ; k = x@Wk^T ; v = x@Wv^T
//   out = softmax(q k^T / sqrt(d)) v
// N = 8192, D = 1024. All matmuls in bf16 WMMA with f32 accumulate.
// ---------------------------------------------------------------------------

#define N_TOK   8192
#define D_DIM   1024

typedef __bf16 bf16_t;
typedef __attribute__((ext_vector_type(16))) __bf16 v16bf;
typedef __attribute__((ext_vector_type(8)))  __bf16 v8bf;
typedef __attribute__((ext_vector_type(8)))  float  v8f;

// Scheduling-pipeline control: keep loads grouped ahead of WMMAs so waits
// become partial (s_wait_loadcnt N) instead of full drains per WMMA.
//   mask 0x020 = VMEM read, 0x008 = MFMA/WMMA
#if __has_builtin(__builtin_amdgcn_sched_group_barrier)
#define SGB(mask, size, id) __builtin_amdgcn_sched_group_barrier(mask, size, id)
#else
#define SGB(mask, size, id)
#endif

// ---------------- WMMA wrapper ---------------------------------------------
__device__ __forceinline__ v8f wmma_bf16(v16bf a, v16bf b, v8f c) {
    // (neg_a, A, neg_b, B, c_mod, C, reuse_a, reuse_b)
    return __builtin_amdgcn_wmma_f32_16x16x32_bf16(false, a, false, b,
                                                   (short)0, c, false, false);
}

// ---------------- fragment loaders -----------------------------------------
// A-matrix 16x32 bf16 (ISA 7.12.2): lane l -> row (l&15);
//   VGPR0..3 hold K = 8*(l>>4) .. +7, VGPR4..7 hold K = 16 + 8*(l>>4) .. +7.
__device__ __forceinline__ v16bf load_a_frag(const bf16_t* base, int ld,
                                             int row0, int k0, int lane) {
    int m = lane & 15, g = lane >> 4;
    const bf16_t* p0 = base + (size_t)(row0 + m) * ld + k0 + g * 8;
    v8bf lo = *(const v8bf*)p0;
    v8bf hi = *(const v8bf*)(p0 + 16);
    v16bf r;
#pragma unroll
    for (int i = 0; i < 8; ++i) { r[i] = lo[i]; r[i + 8] = hi[i]; }
    return r;
}

// B-matrix 32x16 bf16: lane l -> column (l&15);
//   VGPR0..3 hold K = 8*(l>>4) .. +7, VGPR4..7 hold K = 16 + 8*(l>>4) .. +7.
// Source is "column major ready": row n of `base` supplies column n of B.
__device__ __forceinline__ v16bf load_b_frag(const bf16_t* base, int ld,
                                             int col0, int k0, int lane) {
    int n = lane & 15, g = lane >> 4;
    const bf16_t* p = base + (size_t)(col0 + n) * ld + k0 + g * 8;
    v8bf lo = *(const v8bf*)p;
    v8bf hi = *(const v8bf*)(p + 16);
    v16bf r;
#pragma unroll
    for (int i = 0; i < 8; ++i) { r[i] = lo[i]; r[i + 8] = hi[i]; }
    return r;
}

// ---------------- fp32 -> bf16 conversion ----------------------------------
__global__ void cvt_f32_bf16(const float* __restrict__ src,
                             bf16_t* __restrict__ dst, int n) {
    int i = blockIdx.x * blockDim.x + threadIdx.x;
    if (i < n) dst[i] = (bf16_t)src[i];
}

// ---------------- projection GEMM: Out = X @ W^T ---------------------------
// X:[N,D] bf16 row-major, W:[D,D] bf16 row-major (torch Linear layout).
// Each wave computes a 16x128 tile; block = 8 waves = full 1024 cols of one
// 16-row strip. transposeStore=1 writes Out^T (for V -> Vt[D][N]).
// __launch_bounds__(256, 1): raise the VGPR budget so all 18 in-flight
// fragments + 8 accumulators stay in registers (no scratch spills); we trade
// occupancy for deep intra-wave memory-level parallelism.
__global__ __launch_bounds__(256, 1) void
gemm_xwT(const bf16_t* __restrict__ X,
         const bf16_t* __restrict__ W,
         bf16_t* __restrict__ Out, int transposeStore) {
    const int wave = threadIdx.x >> 5;
    const int lane = threadIdx.x & 31;
    const int m0 = blockIdx.x * 16;
    const int n0 = wave * 128;

    v8f acc[8];
#pragma unroll
    for (int t = 0; t < 8; ++t) acc[t] = (v8f){};

    for (int k0 = 0; k0 < D_DIM; k0 += 32) {
        v16bf a = load_a_frag(X, D_DIM, m0, k0, lane);
        v16bf b[8];
#pragma unroll
        for (int t = 0; t < 8; ++t)
            b[t] = load_b_frag(W, D_DIM, n0 + t * 16, k0, lane);
#pragma unroll
        for (int t = 0; t < 8; ++t)
            acc[t] = wmma_bf16(a, b[t], acc[t]);
        // pipeline: 18 VMEM reads (2 A + 16 B) then 8 WMMAs
        SGB(0x020, 18, 0);
        SGB(0x008, 8, 0);
    }

    const int mrow = 8 * (lane >> 4);
    const int ncol = lane & 15;
#pragma unroll
    for (int t = 0; t < 8; ++t)
#pragma unroll
        for (int v = 0; v < 8; ++v) {
            bf16_t h = (bf16_t)acc[t][v];
            int m = m0 + mrow + v;
            int n = n0 + t * 16 + ncol;
            if (transposeStore) Out[(size_t)n * N_TOK + m] = h;
            else                Out[(size_t)m * D_DIM + n] = h;
        }
}

// ---------------- flash attention ------------------------------------------
// Block: 256 threads = 8 waves, owns 16 query rows.
// Wave w owns dims [128w,128w+128) of: the QK^T partial contraction AND the
// output accumulator. Keys processed 32 at a time; partial scores reduced
// across waves in LDS; online softmax; P tile staged in LDS as bf16.
// K and Vt are 16 MB each in bf16 -> fully L2-resident (192 MB), so fragments
// stream straight from global with no LDS staging.
__global__ __launch_bounds__(256, 1) void
flash_attn(const bf16_t* __restrict__ Q,
           const bf16_t* __restrict__ Km,
           const bf16_t* __restrict__ Vt,
           float* __restrict__ Out) {
    __shared__ float  s_part[8][16 * 32];   // per-wave partial scores
    __shared__ float  s_full[16 * 32];      // reduced scores
    __shared__ bf16_t p_buf[8][16 * 32];    // per-wave softmax P tile
    __shared__ float  r_buf[8][16];         // per-wave row rescale factors

    const int wave = threadIdx.x >> 5;
    const int lane = threadIdx.x & 31;
    const int tid  = threadIdx.x;
    const int m0   = blockIdx.x * 16;
    const int d0   = wave * 128;
    const int mrow = 8 * (lane >> 4);
    const int ncol = lane & 15;

    // Q fragments for this wave's 128-dim slice (persist across key loop).
    v16bf qa[4];
#pragma unroll
    for (int j = 0; j < 4; ++j)
        qa[j] = load_a_frag(Q, D_DIM, m0, d0 + 32 * j, lane);

    v8f o_acc[8];
#pragma unroll
    for (int t = 0; t < 8; ++t) o_acc[t] = (v8f){};

    // online softmax state, one row per lane (lanes 0..15 only)
    float row_m = -1e30f;
    float row_l = 0.0f;

    for (int nb = 0; nb < N_TOK; nb += 32) {
        // ---- partial scores: 16 queries x 32 keys over this wave's dims ---
        v16bf kb[8];
#pragma unroll
        for (int t = 0; t < 2; ++t)
#pragma unroll
            for (int j = 0; j < 4; ++j)
                kb[t * 4 + j] =
                    load_b_frag(Km, D_DIM, nb + 16 * t, d0 + 32 * j, lane);

        v8f sc[2];
        sc[0] = (v8f){}; sc[1] = (v8f){};
#pragma unroll
        for (int t = 0; t < 2; ++t)
#pragma unroll
            for (int j = 0; j < 4; ++j)
                sc[t] = wmma_bf16(qa[j], kb[t * 4 + j], sc[t]);
        // pipeline: 16 VMEM reads then 8 WMMAs
        SGB(0x020, 16, 0);
        SGB(0x008, 8, 0);

#pragma unroll
        for (int t = 0; t < 2; ++t)
#pragma unroll
            for (int v = 0; v < 8; ++v)
                s_part[wave][(mrow + v) * 32 + t * 16 + ncol] = sc[t][v];
        __syncthreads();

        // ---- cross-wave reduction + 1/sqrt(d) scale -----------------------
        for (int e = tid; e < 512; e += 256) {
            float s = 0.0f;
#pragma unroll
            for (int w = 0; w < 8; ++w) s += s_part[w][e];
            s_full[e] = s * 0.03125f;   // 1/sqrt(1024)
        }
        __syncthreads();

        // ---- online softmax (redundant per wave; lanes 0..15, 1 row each) -
        if (lane < 16) {
            const int row = lane;
            float mx = row_m;
#pragma unroll 8
            for (int j = 0; j < 32; ++j) mx = fmaxf(mx, s_full[row * 32 + j]);
            float r = __expf(row_m - mx);
            float sum = 0.0f;
#pragma unroll 8
            for (int j = 0; j < 32; ++j) {
                float p = __expf(s_full[row * 32 + j] - mx);
                sum += p;
                p_buf[wave][row * 32 + j] = (bf16_t)p;
            }
            row_l = row_l * r + sum;
            row_m = mx;
            r_buf[wave][row] = r;
        }
        // same-wave LDS RAW; no workgroup barrier needed here

        // ---- O += P @ V  (preload all 8 V fragments, then WMMAs) ----------
        v16bf pa = load_a_frag(&p_buf[wave][0], 32, 0, 0, lane);
        v16bf vb[8];
#pragma unroll
        for (int t = 0; t < 8; ++t)
            vb[t] = load_b_frag(Vt, N_TOK, d0 + 16 * t, nb, lane);

        // ---- rescale accumulators while V loads are in flight -------------
#pragma unroll
        for (int v = 0; v < 8; ++v) {
            float rr = r_buf[wave][mrow + v];
#pragma unroll
            for (int t = 0; t < 8; ++t) o_acc[t][v] *= rr;
        }

#pragma unroll
        for (int t = 0; t < 8; ++t)
            o_acc[t] = wmma_bf16(pa, vb[t], o_acc[t]);
        // pipeline: 16 VMEM reads then 8 WMMAs (DS/VALU float freely)
        SGB(0x020, 16, 0);
        SGB(0x008, 8, 0);

        __syncthreads();   // protect s_part/s_full for next iteration
    }

    // ---- normalize by row sums and store fp32 -----------------------------
    if (lane < 16) r_buf[wave][lane] = 1.0f / row_l;
#pragma unroll
    for (int v = 0; v < 8; ++v) {
        float inv = r_buf[wave][mrow + v];
#pragma unroll
        for (int t = 0; t < 8; ++t)
            Out[(size_t)(m0 + mrow + v) * D_DIM + d0 + t * 16 + ncol] =
                o_acc[t][v] * inv;
    }
}

// ---------------------------------------------------------------------------
extern "C" void kernel_launch(void* const* d_in, const int* in_sizes, int n_in,
                              void* d_out, int out_size, void* d_ws, size_t ws_size,
                              hipStream_t stream) {
    const float* X  = (const float*)d_in[0];
    const float* Wq = (const float*)d_in[1];
    const float* Wk = (const float*)d_in[2];
    const float* Wv = (const float*)d_in[3];
    float* Out = (float*)d_out;

    // workspace layout (bf16): X | Wq | Wk | Wv | Q | K | Vt  => ~70 MB
    char* ws = (char*)d_ws;
    const size_t SZ_X = (size_t)N_TOK * D_DIM * sizeof(bf16_t);   // 16 MB
    const size_t SZ_W = (size_t)D_DIM * D_DIM * sizeof(bf16_t);   // 2 MB
    bf16_t* Xb  = (bf16_t*)(ws);
    bf16_t* Wqb = (bf16_t*)(ws + SZ_X);
    bf16_t* Wkb = (bf16_t*)(ws + SZ_X + SZ_W);
    bf16_t* Wvb = (bf16_t*)(ws + SZ_X + 2 * SZ_W);
    bf16_t* Qb  = (bf16_t*)(ws + SZ_X + 3 * SZ_W);
    bf16_t* Kb  = (bf16_t*)(ws + 2 * SZ_X + 3 * SZ_W);
    bf16_t* Vtb = (bf16_t*)(ws + 3 * SZ_X + 3 * SZ_W);

    const int nX = N_TOK * D_DIM;
    const int nW = D_DIM * D_DIM;
    cvt_f32_bf16<<<(nX + 255) / 256, 256, 0, stream>>>(X,  Xb,  nX);
    cvt_f32_bf16<<<(nW + 255) / 256, 256, 0, stream>>>(Wq, Wqb, nW);
    cvt_f32_bf16<<<(nW + 255) / 256, 256, 0, stream>>>(Wk, Wkb, nW);
    cvt_f32_bf16<<<(nW + 255) / 256, 256, 0, stream>>>(Wv, Wvb, nW);

    dim3 gblk(N_TOK / 16);     // 512 blocks x 8 waves, each wave 16x128 tile
    gemm_xwT<<<gblk, 256, 0, stream>>>(Xb, Wqb, Qb,  0);
    gemm_xwT<<<gblk, 256, 0, stream>>>(Xb, Wkb, Kb,  0);
    gemm_xwT<<<gblk, 256, 0, stream>>>(Xb, Wvb, Vtb, 1);   // store V^T

    flash_attn<<<dim3(N_TOK / 16), 256, 0, stream>>>(Qb, Kb, Vtb, Out);
}
